// SDPA_24335284699389
// MI455X (gfx1250) — compile-verified, matrix operand order checked by reading, not run on hardware
//
#include <hip/hip_runtime.h>

typedef __attribute__((ext_vector_type(16))) _Float16 v16h;
typedef __attribute__((ext_vector_type(8)))  _Float16 v8h;
typedef __attribute__((ext_vector_type(4)))  _Float16 v4h;
typedef __attribute__((ext_vector_type(2)))  _Float16 v2h;
typedef __attribute__((ext_vector_type(8)))  float    v8f;

#define T_DIM 2048
#define NHEAD 32
#define KVH   8
#define HD    128
#define QTILE 64              // query rows per workgroup
#define SBLK  64              // key block size
#define WAVES 4
#define KROWP 136             // padded row stride (halves) for K tile [64 x 128]
#define VROWP 72              // padded row stride (halves) for Vt [128 x 64]
#define KTILE_SZ (SBLK * KROWP)
#define VTILE_SZ (HD * VROWP)

static __device__ __forceinline__ v16h mk16(v8h lo, v8h hi) {
  return __builtin_shufflevector(lo, hi, 0, 1, 2, 3, 4, 5, 6, 7,
                                         8, 9, 10, 11, 12, 13, 14, 15);
}

// ---------------------------------------------------------------------------
// Pre-pass: fp32 k/v  ->  f16 Kh[kvh][s][h] (row-major) and Vt[kvh][h][s]
// (transposed through an LDS tile so both global sides stay coalesced).
// ---------------------------------------------------------------------------
__global__ __launch_bounds__(256)
void cvt_kv_kernel(const float* __restrict__ k, const float* __restrict__ v,
                   _Float16* __restrict__ Kh, _Float16* __restrict__ Vt) {
  __shared__ alignas(16) _Float16 tile[64 * 136];   // V tile [s][h], padded
  const int tid = threadIdx.x;
  const int s0  = blockIdx.x * 64;
  const int kvh = blockIdx.y;
#pragma unroll
  for (int i = 0; i < 16; ++i) {
    const int e  = i * 256 + tid;      // pair id over 64 s x 64 h-pairs
    const int s  = e >> 6;             // 0..63
    const int h2 = (e & 63) * 2;       // even h
    const size_t src = ((size_t)(s0 + s) * KVH + kvh) * HD + h2;
    const float2 kk = *(const float2*)(k + src);
    const float2 vv = *(const float2*)(v + src);
    v2h kh; kh[0] = (_Float16)kk.x; kh[1] = (_Float16)kk.y;
    *(v2h*)(Kh + ((size_t)kvh * T_DIM + s0 + s) * HD + h2) = kh;
    v2h vh; vh[0] = (_Float16)vv.x; vh[1] = (_Float16)vv.y;
    *(v2h*)(tile + s * 136 + h2) = vh;
  }
  __syncthreads();
#pragma unroll
  for (int i = 0; i < 16; ++i) {
    const int e  = i * 256 + tid;      // pair id over 128 h x 32 s-pairs
    const int h  = e >> 5;             // 0..127
    const int s2 = (e & 31) * 2;       // even s
    v2h o2;
    o2[0] = tile[s2 * 136 + h];
    o2[1] = tile[(s2 + 1) * 136 + h];
    *(v2h*)(Vt + ((size_t)kvh * HD + h) * T_DIM + s0 + s2) = o2;
  }
}

// ---------------------------------------------------------------------------
// Fast path: flash attention with pre-converted f16 K / transposed V.
// ---------------------------------------------------------------------------
__global__ __launch_bounds__(128)
void sdpa_fa16_kernel(const float* __restrict__ q,
                      const _Float16* __restrict__ Kh,
                      const _Float16* __restrict__ Vt,
                      float* __restrict__ out) {
  __shared__ alignas(16) _Float16 sK[2 * KTILE_SZ];
  __shared__ alignas(16) _Float16 sVt[2 * VTILE_SZ];

  const int tid  = threadIdx.x;
  const int wave = tid >> 5;
  const int lane = tid & 31;
  const int ln   = lane & 15;
  const int g    = lane >> 4;

  const int qbase = blockIdx.x * QTILE;
  const int head  = blockIdx.y;
  const int kvh   = head >> 2;

  const float qscale = 0.1275174306f;          // log2(e)/sqrt(128)
  const int qr = qbase + wave * 16 + ln;
  const float* qp = q + ((size_t)qr * NHEAD + head) * HD;
  v16h bq[4];
#pragma unroll
  for (int c = 0; c < 4; ++c) {
#pragma unroll
    for (int j = 0; j < 8; ++j) {
      bq[c][j]     = (_Float16)(qp[32 * c + 8 * g + j] * qscale);
      bq[c][8 + j] = (_Float16)(qp[32 * c + 16 + 8 * g + j] * qscale);
    }
  }

  v8f o[8];
#pragma unroll
  for (int c = 0; c < 8; ++c) o[c] = (v8f){0.f, 0.f, 0.f, 0.f, 0.f, 0.f, 0.f, 0.f};
  float mi = -1e30f, li = 0.f;

  const int nblk = blockIdx.x + 1;

  auto stage = [&](int sk, int b) {   // pure f16 b128 copies, no conversion
    const _Float16* kp = Kh + ((size_t)kvh * T_DIM + sk) * HD;
    const _Float16* vp = Vt + (size_t)kvh * HD * T_DIM + sk;
    _Float16* dK = sK + b * KTILE_SZ;
    _Float16* dV = sVt + b * VTILE_SZ;
#pragma unroll
    for (int i = 0; i < 8; ++i) {
      const int e = i * 128 + tid;     // K tile: 64 rows x 16 chunks of 16B
      const int s = e >> 4, c = e & 15;
      *(v8h*)(dK + s * KROWP + c * 8) = *(const v8h*)(kp + s * HD + c * 8);
    }
#pragma unroll
    for (int i = 0; i < 8; ++i) {
      const int e = i * 128 + tid;     // Vt tile: 128 rows x 8 chunks of 16B
      const int h = e >> 3, c = e & 7;
      *(v8h*)(dV + h * VROWP + c * 8) = *(const v8h*)(vp + (size_t)h * T_DIM + c * 8);
    }
  };

  stage(0, 0);
  __syncthreads();
  int buf = 0;

  for (int kb = 0; kb < nblk; ++kb) {
    const int sk = kb * SBLK;
    if (kb + 1 < nblk) stage(sk + SBLK, buf ^ 1);

    const _Float16* tK = sK + buf * KTILE_SZ;
    const _Float16* tV = sVt + buf * VTILE_SZ;

    v8f sc[4];
#pragma unroll
    for (int t = 0; t < 4; ++t) sc[t] = (v8f){0.f, 0.f, 0.f, 0.f, 0.f, 0.f, 0.f, 0.f};
#pragma unroll
    for (int c = 0; c < 4; ++c) {
#pragma unroll
      for (int t = 0; t < 4; ++t) {
        const _Float16* ka = tK + (16 * t + ln) * KROWP + 32 * c + 8 * g;
        const v16h a = mk16(*(const v8h*)ka, *(const v8h*)(ka + 16));
        sc[t] = __builtin_amdgcn_wmma_f32_16x16x32_f16(false, a, false, bq[c],
                                                       (short)0, sc[t], false, false);
      }
    }

    if (kb == nblk - 1) {
      const int lim0 = qr - sk - 8 * g;
#pragma unroll
      for (int t = 0; t < 4; ++t) {
#pragma unroll
        for (int r = 0; r < 8; ++r)
          sc[t][r] = (16 * t + r <= lim0) ? sc[t][r] : -1e30f;
      }
    }

    float rmx = sc[0][0];
#pragma unroll
    for (int t = 0; t < 4; ++t) {
#pragma unroll
      for (int r = 0; r < 8; ++r) rmx = fmaxf(rmx, sc[t][r]);
    }
    rmx = fmaxf(rmx, __shfl_xor(rmx, 16, 32));

    if (__any(rmx > mi)) {
      const float mnew  = fmaxf(mi, rmx);
      const float alpha = __builtin_amdgcn_exp2f(mi - mnew);
      mi = mnew;
      li *= alpha;
#pragma unroll
      for (int c = 0; c < 8; ++c) {
#pragma unroll
        for (int r = 0; r < 8; ++r) o[c][r] *= alpha;
      }
    }

    v16h pb0, pb1;
    float rs = 0.f;
#pragma unroll
    for (int r = 0; r < 8; ++r) {
      const float p0 = __builtin_amdgcn_exp2f(sc[0][r] - mi);
      const float p1 = __builtin_amdgcn_exp2f(sc[1][r] - mi);
      const float p2 = __builtin_amdgcn_exp2f(sc[2][r] - mi);
      const float p3 = __builtin_amdgcn_exp2f(sc[3][r] - mi);
      rs += (p0 + p1) + (p2 + p3);
      pb0[r]     = (_Float16)p0;
      pb0[8 + r] = (_Float16)p1;
      pb1[r]     = (_Float16)p2;
      pb1[8 + r] = (_Float16)p3;
    }
    rs += __shfl_xor(rs, 16, 32);
    li += rs;

#pragma unroll
    for (int c = 0; c < 8; ++c) {
      const _Float16* va = tV + (16 * c + ln) * VROWP + 8 * g;
      const v16h a0 = mk16(*(const v8h*)va, *(const v8h*)(va + 16));
      o[c] = __builtin_amdgcn_wmma_f32_16x16x32_f16(false, a0, false, pb0,
                                                    (short)0, o[c], false, false);
      const v16h a1 = mk16(*(const v8h*)(va + 32), *(const v8h*)(va + 48));
      o[c] = __builtin_amdgcn_wmma_f32_16x16x32_f16(false, a1, false, pb1,
                                                    (short)0, o[c], false, false);
    }

    __syncthreads();
    buf ^= 1;
  }

  const float inv = 1.0f / li;
  float* op = out + ((size_t)qr * NHEAD + head) * HD + 8 * g;
#pragma unroll
  for (int c = 0; c < 8; ++c) {
    float4 lo = { o[c][0] * inv, o[c][1] * inv, o[c][2] * inv, o[c][3] * inv };
    float4 hi = { o[c][4] * inv, o[c][5] * inv, o[c][6] * inv, o[c][7] * inv };
    *(float4*)(op + 16 * c)     = lo;
    *(float4*)(op + 16 * c + 4) = hi;
  }
}

// ---------------------------------------------------------------------------
// Fallback: self-contained fp32->f16 staging (round-3 kernel), used only if
// the workspace is too small for the f16 pre-pass buffers.
// ---------------------------------------------------------------------------
__global__ __launch_bounds__(128)
void sdpa_fa_kernel(const float* __restrict__ q,
                    const float* __restrict__ k,
                    const float* __restrict__ v,
                    float* __restrict__ out) {
  __shared__ alignas(16) _Float16 sK[2 * KTILE_SZ];
  __shared__ alignas(16) _Float16 sVt[2 * VTILE_SZ];

  const int tid  = threadIdx.x;
  const int wave = tid >> 5;
  const int lane = tid & 31;
  const int ln   = lane & 15;
  const int g    = lane >> 4;

  const int qbase = blockIdx.x * QTILE;
  const int head  = blockIdx.y;
  const int kvh   = head >> 2;

  const float qscale = 0.1275174306f;
  const int qr = qbase + wave * 16 + ln;
  const float* qp = q + ((size_t)qr * NHEAD + head) * HD;
  v16h bq[4];
#pragma unroll
  for (int c = 0; c < 4; ++c) {
#pragma unroll
    for (int j = 0; j < 8; ++j) {
      bq[c][j]     = (_Float16)(qp[32 * c + 8 * g + j] * qscale);
      bq[c][8 + j] = (_Float16)(qp[32 * c + 16 + 8 * g + j] * qscale);
    }
  }

  v8f o[8];
#pragma unroll
  for (int c = 0; c < 8; ++c) o[c] = (v8f){0.f, 0.f, 0.f, 0.f, 0.f, 0.f, 0.f, 0.f};
  float mi = -1e30f, li = 0.f;

  const int nblk = blockIdx.x + 1;

  auto stage = [&](int sk, int b) {
    const float* kp = k + ((size_t)sk * KVH + kvh) * HD;
    const float* vp = v + ((size_t)sk * KVH + kvh) * HD;
    _Float16* dK = sK + b * KTILE_SZ;
    _Float16* dV = sVt + b * VTILE_SZ;
#pragma unroll
    for (int i = 0; i < 16; ++i) {
      const int s = i * 4 + wave;
      const int h = lane * 4;
      const float4 k4 = *(const float4*)(kp + (size_t)s * (KVH * HD) + h);
      v4h kh;
      kh[0] = (_Float16)k4.x; kh[1] = (_Float16)k4.y;
      kh[2] = (_Float16)k4.z; kh[3] = (_Float16)k4.w;
      *(v4h*)(dK + s * KROWP + h) = kh;
    }
#pragma unroll
    for (int jj = 0; jj < 4; ++jj) {
      const int h = lane + 32 * jj;
#pragma unroll
      for (int si = 0; si < 8; ++si) {
        const int s0 = si * 8 + wave * 2;
        const float va = vp[(size_t)s0 * (KVH * HD) + h];
        const float vb = vp[(size_t)(s0 + 1) * (KVH * HD) + h];
        v2h vh; vh[0] = (_Float16)va; vh[1] = (_Float16)vb;
        *(v2h*)(dV + h * VROWP + s0) = vh;
      }
    }
  };

  stage(0, 0);
  __syncthreads();
  int buf = 0;

  for (int kb = 0; kb < nblk; ++kb) {
    const int sk = kb * SBLK;
    if (kb + 1 < nblk) stage(sk + SBLK, buf ^ 1);

    const _Float16* tK = sK + buf * KTILE_SZ;
    const _Float16* tV = sVt + buf * VTILE_SZ;

    v8f sc[4];
#pragma unroll
    for (int t = 0; t < 4; ++t) sc[t] = (v8f){0.f, 0.f, 0.f, 0.f, 0.f, 0.f, 0.f, 0.f};
#pragma unroll
    for (int c = 0; c < 4; ++c) {
#pragma unroll
      for (int t = 0; t < 4; ++t) {
        const _Float16* ka = tK + (16 * t + ln) * KROWP + 32 * c + 8 * g;
        const v16h a = mk16(*(const v8h*)ka, *(const v8h*)(ka + 16));
        sc[t] = __builtin_amdgcn_wmma_f32_16x16x32_f16(false, a, false, bq[c],
                                                       (short)0, sc[t], false, false);
      }
    }

    if (kb == nblk - 1) {
      const int lim0 = qr - sk - 8 * g;
#pragma unroll
      for (int t = 0; t < 4; ++t) {
#pragma unroll
        for (int r = 0; r < 8; ++r)
          sc[t][r] = (16 * t + r <= lim0) ? sc[t][r] : -1e30f;
      }
    }

    float rmx = sc[0][0];
#pragma unroll
    for (int t = 0; t < 4; ++t) {
#pragma unroll
      for (int r = 0; r < 8; ++r) rmx = fmaxf(rmx, sc[t][r]);
    }
    rmx = fmaxf(rmx, __shfl_xor(rmx, 16, 32));

    if (__any(rmx > mi)) {
      const float mnew  = fmaxf(mi, rmx);
      const float alpha = __builtin_amdgcn_exp2f(mi - mnew);
      mi = mnew;
      li *= alpha;
#pragma unroll
      for (int c = 0; c < 8; ++c) {
#pragma unroll
        for (int r = 0; r < 8; ++r) o[c][r] *= alpha;
      }
    }

    v16h pb0, pb1;
    float rs = 0.f;
#pragma unroll
    for (int r = 0; r < 8; ++r) {
      const float p0 = __builtin_amdgcn_exp2f(sc[0][r] - mi);
      const float p1 = __builtin_amdgcn_exp2f(sc[1][r] - mi);
      const float p2 = __builtin_amdgcn_exp2f(sc[2][r] - mi);
      const float p3 = __builtin_amdgcn_exp2f(sc[3][r] - mi);
      rs += (p0 + p1) + (p2 + p3);
      pb0[r]     = (_Float16)p0;
      pb0[8 + r] = (_Float16)p1;
      pb1[r]     = (_Float16)p2;
      pb1[8 + r] = (_Float16)p3;
    }
    rs += __shfl_xor(rs, 16, 32);
    li += rs;

#pragma unroll
    for (int c = 0; c < 8; ++c) {
      const _Float16* va = tV + (16 * c + ln) * VROWP + 8 * g;
      const v16h a0 = mk16(*(const v8h*)va, *(const v8h*)(va + 16));
      o[c] = __builtin_amdgcn_wmma_f32_16x16x32_f16(false, a0, false, pb0,
                                                    (short)0, o[c], false, false);
      const v16h a1 = mk16(*(const v8h*)(va + 32), *(const v8h*)(va + 48));
      o[c] = __builtin_amdgcn_wmma_f32_16x16x32_f16(false, a1, false, pb1,
                                                    (short)0, o[c], false, false);
    }

    __syncthreads();
    buf ^= 1;
  }

  const float inv = 1.0f / li;
  float* op = out + ((size_t)qr * NHEAD + head) * HD + 8 * g;
#pragma unroll
  for (int c = 0; c < 8; ++c) {
    float4 lo = { o[c][0] * inv, o[c][1] * inv, o[c][2] * inv, o[c][3] * inv };
    float4 hi = { o[c][4] * inv, o[c][5] * inv, o[c][6] * inv, o[c][7] * inv };
    *(float4*)(op + 16 * c)     = lo;
    *(float4*)(op + 16 * c + 4) = hi;
  }
}

extern "C" void kernel_launch(void* const* d_in, const int* in_sizes, int n_in,
                              void* d_out, int out_size, void* d_ws, size_t ws_size,
                              hipStream_t stream) {
  (void)in_sizes; (void)n_in; (void)out_size;
  const float* q = (const float*)d_in[0];
  const float* k = (const float*)d_in[1];
  const float* v = (const float*)d_in[2];
  float* out = (float*)d_out;
  dim3 grid(T_DIM / QTILE, NHEAD, 1);
  dim3 block(WAVES * 32, 1, 1);

  const size_t kv_halves = (size_t)KVH * T_DIM * HD;        // 2M halves each
  const size_t need = 2 * kv_halves * sizeof(_Float16);     // 8 MiB
  if (d_ws != nullptr && ws_size >= need) {
    _Float16* Kh = (_Float16*)d_ws;
    _Float16* Vt = Kh + kv_halves;
    cvt_kv_kernel<<<dim3(T_DIM / 64, KVH, 1), dim3(256, 1, 1), 0, stream>>>(k, v, Kh, Vt);
    sdpa_fa16_kernel<<<grid, block, 0, stream>>>(q, Kh, Vt, out);
  } else {
    sdpa_fa_kernel<<<grid, block, 0, stream>>>(q, k, v, out);
  }
}
